// SNN_72413148611059
// MI455X (gfx1250) — compile-verified, matrix operand order checked by reading, not run on hardware
//
#include <hip/hip_runtime.h>

typedef _Float16 v16h __attribute__((ext_vector_type(16)));
typedef _Float16 v8h  __attribute__((ext_vector_type(8)));
typedef float    v8f  __attribute__((ext_vector_type(8)));

#define B_     256
#define T_     32
#define IN_    2312
#define KP1_   2336   // 73*32, K padded for layer-1 GEMM
#define H_     1024
#define OUT_   10
#define OUTP_  16

// LIF constants: DT*TAU_MEM_INV = 1e-3*50 = 0.05 ; i decay = 1 - 1e-3*200 = 0.8
#define DTM_   0.05f
#define IDEC_  0.8f
#define VTH_   1.0f

__device__ __forceinline__ float lif_update(float inp, float& v, float& i) {
    float vdec = v + DTM_ * (i - v);
    float idec = IDEC_ * i;
    float s = (vdec > VTH_) ? 1.0f : 0.0f;
    v = (1.0f - s) * vdec;
    i = idec + inp;
    return s;
}

// CDNA5 async DMA: global -> LDS, 16 bytes per lane, tracked by ASYNCcnt.
// Low 32 bits of a generic pointer to __shared__ == wave-relative LDS address.
__device__ __forceinline__ void async_copy_b128(void* lds_dst, const void* gsrc) {
    unsigned int       loff  = (unsigned int)(unsigned long long)(uintptr_t)lds_dst;
    unsigned long long gaddr = (unsigned long long)(uintptr_t)gsrc;
    asm volatile("global_load_async_to_lds_b128 %0, %1, off"
                 :: "v"(loff), "v"(gaddr)
                 : "memory");
}
__device__ __forceinline__ void wait_async0() {
    asm volatile("s_wait_asynccnt 0x0" ::: "memory");
}

// Load one 16x32 f16 WMMA A/B fragment from an LDS tile stored row-major
// with row stride 32 halves. Lane L: row = base + (L&15); elements 0..7 hold
// K = (L>>4)*8 + 0..7, elements 8..15 hold K = 16 + (L>>4)*8 + 0..7.
__device__ __forceinline__ v16h frag_from_lds(const _Float16* tile, int rowBase, int lane) {
    int r = rowBase + (lane & 15);
    int h = (lane >> 4) << 3;                  // 0 or 8
    const _Float16* p = tile + r * 32 + h;     // 16B aligned
    v16h f;
    *(v8h*)&f       = *(const v8h*)(p);        // K = h .. h+7
    *((v8h*)&f + 1) = *(const v8h*)(p + 16);   // K = 16+h .. 16+h+7
    return f;
}

// ---------------------------------------------------------------------------
// Weight conversion kernels (run once per launch)
// ---------------------------------------------------------------------------
__global__ void conv_w1(const float* __restrict__ W1, _Float16* __restrict__ W1h) {
    int idx = blockIdx.x * 256 + threadIdx.x;            // over 1024*2336
    if (idx >= H_ * KP1_) return;
    int n = idx / KP1_, k = idx - n * KP1_;
    W1h[idx] = (k < IN_) ? (_Float16)W1[(size_t)n * IN_ + k] : (_Float16)0.0f;
}

__global__ void conv_w2(const float* __restrict__ W2, _Float16* __restrict__ W2h) {
    int idx = blockIdx.x * 256 + threadIdx.x;            // over 1024*1024
    if (idx >= H_ * H_) return;
    W2h[idx] = (_Float16)W2[idx];
}

__global__ void conv_wout(const float* __restrict__ Wout, _Float16* __restrict__ Wouth) {
    int idx = blockIdx.x * 256 + threadIdx.x;            // over 16*1024
    if (idx >= OUTP_ * H_) return;
    int n = idx >> 10, k = idx & (H_ - 1);
    Wouth[idx] = (n < OUT_) ? (_Float16)Wout[(size_t)n * H_ + k] : (_Float16)0.0f;
}

__global__ void init_state(float* v1, float* i1, float* v2, float* i2,
                           float* vo, float* io, float* out) {
    int idx = blockIdx.x * 256 + threadIdx.x;            // 262144 threads
    if (idx < B_ * H_) { v1[idx] = 0.f; i1[idx] = 0.f; v2[idx] = 0.f; i2[idx] = 0.f; }
    if (idx < B_ * OUTP_) { vo[idx] = 0.f; io[idx] = 0.f; }
    if (idx < B_ * OUT_) out[idx] = 0.f;
}

// ---------------------------------------------------------------------------
// GEMM1: Z1[8192,1024] = Xf32[8192,2312] (cvt f16) @ W1h[1024,2336]^T
// Tile 128x128, 8 waves (2x4), wave tile 64x32 = 4x2 WMMA accumulators.
// Double-buffered LDS; B side uses async global->LDS DMA, A side converts
// f32->f16 in registers.
// ---------------------------------------------------------------------------
__global__ __launch_bounds__(256) void gemm1_kernel(
    const float* __restrict__ x, const _Float16* __restrict__ W1h,
    float* __restrict__ Z1)
{
    __shared__ __align__(16) _Float16 As[2][128 * 32];
    __shared__ __align__(16) _Float16 Bs[2][128 * 32];

    const int m0 = blockIdx.x * 128;       // 64 row tiles
    const int n0 = blockIdx.y * 128;       // 8  col tiles
    const int tid = threadIdx.x;
    const int lane = tid & 31;
    const int wave = tid >> 5;
    const int wr = (wave >> 2) * 64;       // wave row base within tile
    const int wc = (wave & 3) * 32;        // wave col base within tile
    const int lr = tid >> 2;               // loader row 0..63 (+64 on 2nd chunk)
    const int lk = (tid & 3) * 8;          // loader k offset within tile

    v8f acc[4][2] = {};

    auto load_tiles = [&](int kt, int buf) {
        #pragma unroll
        for (int it = 0; it < 2; ++it) {
            int r = lr + it * 64;
            // A: manual f32 load + convert (IN_ % 8 == 0 -> group all-valid)
            _Float16* dst = &As[buf][r * 32 + lk];
            if (kt + lk < IN_) {
                v8f fv = *(const v8f*)(x + (size_t)(m0 + r) * IN_ + kt + lk);
                *(v8h*)dst = __builtin_convertvector(fv, v8h);
            } else {
                v8h z = {};
                *(v8h*)dst = z;
            }
            // B: async DMA (W1h is K-padded, always in range)
            async_copy_b128(&Bs[buf][r * 32 + lk],
                            W1h + (size_t)(n0 + r) * KP1_ + kt + lk);
        }
    };

    auto compute = [&](int buf) {
        v16h af[4], bf[2];
        #pragma unroll
        for (int i = 0; i < 4; ++i) af[i] = frag_from_lds(&As[buf][0], wr + i * 16, lane);
        #pragma unroll
        for (int j = 0; j < 2; ++j) bf[j] = frag_from_lds(&Bs[buf][0], wc + j * 16, lane);
        #pragma unroll
        for (int i = 0; i < 4; ++i)
            #pragma unroll
            for (int j = 0; j < 2; ++j)
                acc[i][j] = __builtin_amdgcn_wmma_f32_16x16x32_f16(
                    false, af[i], false, bf[j], (short)0, acc[i][j], false, false);
    };

    constexpr int NK = KP1_ / 32;          // 73
    load_tiles(0, 0);
    wait_async0();
    __syncthreads();
    for (int k = 0; k < NK; ++k) {
        int cur = k & 1;
        if (k + 1 < NK) load_tiles((k + 1) * 32, cur ^ 1);
        compute(cur);
        wait_async0();
        __syncthreads();
    }

    const int nl = lane & 15;
    const int ml = (lane >> 4) * 8;
    #pragma unroll
    for (int i = 0; i < 4; ++i)
        #pragma unroll
        for (int j = 0; j < 2; ++j) {
            int gm = m0 + wr + i * 16 + ml;
            int gn = n0 + wc + j * 16 + nl;
            #pragma unroll
            for (int v = 0; v < 8; ++v)
                Z1[(size_t)(gm + v) * H_ + gn] = acc[i][j][v];
        }
}

// ---------------------------------------------------------------------------
// LIF layer 1 (elementwise over B x H) for timestep t; emits f16 spikes.
// Z1 rows are laid out as m = b*T + t (matching x's (B,T,IN) memory order).
// ---------------------------------------------------------------------------
__global__ __launch_bounds__(256) void lif1_kernel(
    const float* __restrict__ Z1, const float* __restrict__ b1,
    float* __restrict__ v1, float* __restrict__ i1,
    _Float16* __restrict__ s1h, int t)
{
    int idx = blockIdx.x * 256 + threadIdx.x;   // 0 .. 262143
    int b = idx >> 10, n = idx & (H_ - 1);
    float inp = Z1[((size_t)b * T_ + t) * H_ + n] + b1[n];
    float v = v1[idx], cur = i1[idx];
    float s = lif_update(inp, v, cur);
    v1[idx] = v; i1[idx] = cur;
    s1h[idx] = (_Float16)s;
}

// ---------------------------------------------------------------------------
// GEMM2 + fused LIF2: inp2[256,1024] = s1h @ W2h^T + b2, then LIF in the
// epilogue. Tile 64x64, 4 waves (2x2), wave tile 32x32 = 2x2 accumulators.
// Grid (4,16) = 64 workgroups for better WGP coverage on the small GEMM.
// Both A and B staged via async global->LDS DMA, double buffered.
// ---------------------------------------------------------------------------
__global__ __launch_bounds__(128) void gemm2_lif_kernel(
    const _Float16* __restrict__ s1h, const _Float16* __restrict__ W2h,
    const float* __restrict__ b2,
    float* __restrict__ v2, float* __restrict__ i2,
    _Float16* __restrict__ s2h)
{
    __shared__ __align__(16) _Float16 As[2][64 * 32];
    __shared__ __align__(16) _Float16 Bs[2][64 * 32];

    const int m0 = blockIdx.x * 64;    // 4 row tiles (B=256)
    const int n0 = blockIdx.y * 64;    // 16 col tiles (H=1024)
    const int tid = threadIdx.x;
    const int lane = tid & 31;
    const int wave = tid >> 5;         // 0..3
    const int wr = (wave >> 1) * 32;
    const int wc = (wave & 1) * 32;
    const int lr = tid >> 2;           // loader row 0..31 (+32 on 2nd chunk)
    const int lk = (tid & 3) * 8;

    v8f acc[2][2] = {};

    auto load_tiles = [&](int kt, int buf) {
        #pragma unroll
        for (int it = 0; it < 2; ++it) {
            int r = lr + it * 32;
            async_copy_b128(&As[buf][r * 32 + lk],
                            s1h + (size_t)(m0 + r) * H_ + kt + lk);
            async_copy_b128(&Bs[buf][r * 32 + lk],
                            W2h + (size_t)(n0 + r) * H_ + kt + lk);
        }
    };

    auto compute = [&](int buf) {
        v16h af[2], bf[2];
        #pragma unroll
        for (int i = 0; i < 2; ++i) af[i] = frag_from_lds(&As[buf][0], wr + i * 16, lane);
        #pragma unroll
        for (int j = 0; j < 2; ++j) bf[j] = frag_from_lds(&Bs[buf][0], wc + j * 16, lane);
        #pragma unroll
        for (int i = 0; i < 2; ++i)
            #pragma unroll
            for (int j = 0; j < 2; ++j)
                acc[i][j] = __builtin_amdgcn_wmma_f32_16x16x32_f16(
                    false, af[i], false, bf[j], (short)0, acc[i][j], false, false);
    };

    constexpr int NK = H_ / 32;        // 32
    load_tiles(0, 0);
    wait_async0();
    __syncthreads();
    for (int k = 0; k < NK; ++k) {
        int cur = k & 1;
        if (k + 1 < NK) load_tiles((k + 1) * 32, cur ^ 1);
        compute(cur);
        wait_async0();
        __syncthreads();
    }

    const int nl = lane & 15;
    const int ml = (lane >> 4) * 8;
    #pragma unroll
    for (int i = 0; i < 2; ++i)
        #pragma unroll
        for (int j = 0; j < 2; ++j) {
            int gm = m0 + wr + i * 16 + ml;
            int gn = n0 + wc + j * 16 + nl;
            float bias = b2[gn];
            #pragma unroll
            for (int v = 0; v < 8; ++v) {
                size_t idx = (size_t)(gm + v) * H_ + gn;
                float inp = acc[i][j][v] + bias;
                float vv = v2[idx], cur = i2[idx];
                float s = lif_update(inp, vv, cur);
                v2[idx] = vv; i2[idx] = cur;
                s2h[idx] = (_Float16)s;
            }
        }
}

// ---------------------------------------------------------------------------
// Output layer: inp_o[256,16] = s2h @ Wouth^T + bout, LIF, rate accumulation.
// 16 blocks x 256 threads; thread = (b = blk*16 + tid/16, o = tid%16).
// ---------------------------------------------------------------------------
__global__ __launch_bounds__(256) void out_kernel(
    const _Float16* __restrict__ s2h, const _Float16* __restrict__ Wouth,
    const float* __restrict__ bout,
    float* __restrict__ vo, float* __restrict__ io, float* __restrict__ out)
{
    int tid = threadIdx.x;
    int b = blockIdx.x * 16 + (tid >> 4);
    int o = tid & 15;
    const _Float16* sr = s2h + (size_t)b * H_;
    const _Float16* wr = Wouth + (size_t)o * H_;
    float acc = 0.0f;
    for (int k = 0; k < H_; k += 8) {
        v8h a = *(const v8h*)(sr + k);
        v8h w = *(const v8h*)(wr + k);
        #pragma unroll
        for (int e = 0; e < 8; ++e) acc += (float)a[e] * (float)w[e];
    }
    float inp = acc + ((o < OUT_) ? bout[o] : 0.0f);
    int idx = b * OUTP_ + o;
    float v = vo[idx], cur = io[idx];
    float s = lif_update(inp, v, cur);
    vo[idx] = v; io[idx] = cur;
    if (o < OUT_) out[b * OUT_ + o] += s;
}

// ---------------------------------------------------------------------------
extern "C" void kernel_launch(void* const* d_in, const int* in_sizes, int n_in,
                              void* d_out, int out_size, void* d_ws, size_t ws_size,
                              hipStream_t stream) {
    const float* x    = (const float*)d_in[0];   // (256,32,2,34,34) = (8192,2312)
    const float* W1   = (const float*)d_in[1];   // (1024,2312)
    const float* b1   = (const float*)d_in[2];   // (1024,)
    const float* W2   = (const float*)d_in[3];   // (1024,1024)
    const float* b2   = (const float*)d_in[4];   // (1024,)
    const float* Wout = (const float*)d_in[5];   // (10,1024)
    const float* bout = (const float*)d_in[6];   // (10,)
    float* out = (float*)d_out;                  // (256,10)

    char* ws = (char*)d_ws;
    _Float16* W1h   = (_Float16*)ws;                  ws += (size_t)H_ * KP1_ * 2;     // 4.78 MB
    _Float16* W2h   = (_Float16*)ws;                  ws += (size_t)H_ * H_ * 2;       // 2 MB
    _Float16* Wouth = (_Float16*)ws;                  ws += (size_t)OUTP_ * H_ * 2;    // 32 KB
    float*    Z1    = (float*)ws;                     ws += (size_t)B_ * T_ * H_ * 4;  // 33.5 MB
    float*    v1    = (float*)ws;                     ws += (size_t)B_ * H_ * 4;
    float*    i1    = (float*)ws;                     ws += (size_t)B_ * H_ * 4;
    float*    v2    = (float*)ws;                     ws += (size_t)B_ * H_ * 4;
    float*    i2    = (float*)ws;                     ws += (size_t)B_ * H_ * 4;
    float*    vo    = (float*)ws;                     ws += (size_t)B_ * OUTP_ * 4;
    float*    io    = (float*)ws;                     ws += (size_t)B_ * OUTP_ * 4;
    _Float16* s1h   = (_Float16*)ws;                  ws += (size_t)B_ * H_ * 2;
    _Float16* s2h   = (_Float16*)ws;                  ws += (size_t)B_ * H_ * 2;

    // One-time (per launch) weight conversion + state init.
    conv_w1<<<(H_ * KP1_ + 255) / 256, 256, 0, stream>>>(W1, W1h);
    conv_w2<<<(H_ * H_ + 255) / 256, 256, 0, stream>>>(W2, W2h);
    conv_wout<<<(OUTP_ * H_ + 255) / 256, 256, 0, stream>>>(Wout, Wouth);
    init_state<<<(B_ * H_ + 255) / 256, 256, 0, stream>>>(v1, i1, v2, i2, vo, io, out);

    // Big batched GEMM for all timesteps of layer-1 currents.
    gemm1_kernel<<<dim3(B_ * T_ / 128, H_ / 128), 256, 0, stream>>>(x, W1h, Z1);

    // Sequential recurrence over T.
    for (int t = 0; t < T_; ++t) {
        lif1_kernel<<<(B_ * H_ + 255) / 256, 256, 0, stream>>>(Z1, b1, v1, i1, s1h, t);
        gemm2_lif_kernel<<<dim3(B_ / 64, H_ / 64), 128, 0, stream>>>(s1h, W2h, b2, v2, i2, s2h);
        out_kernel<<<B_ / 16, 256, 0, stream>>>(s2h, Wouth, bout, vo, io, out);
    }
}
